// MoE_66967130079941
// MI455X (gfx1250) — compile-verified, matrix operand order checked by reading, not run on hardware
//
#include <hip/hip_runtime.h>
#include <hip/hip_bf16.h>
#include <math.h>

// ---------------- problem constants ----------------
#define T_TOK 2048      // S*B tokens
#define DD    1024      // model dim
#define EE    15        // experts
#define KTOP  3
#define FF    1024      // expert FFN inner
#define FSH   2048      // shared FFN inner
#define POOL_CAP 8064   // T*KTOP + EE*128 padding worst case, multiple of 128
#define LDSP  72        // padded LDS row length (bank-conflict-free frag gather)

typedef __attribute__((ext_vector_type(16))) __bf16 v16bf;
typedef __attribute__((ext_vector_type(8)))  float  v8f;

// ---------------- helpers ----------------
__device__ __forceinline__ unsigned short f2bf(float f) {
  unsigned int u = __float_as_uint(f);
  unsigned int r = u + 0x7FFFu + ((u >> 16) & 1u);  // round-to-nearest-even
  return (unsigned short)(r >> 16);
}

__device__ __forceinline__ float gelu_exact(float g) {
  return 0.5f * g * (1.0f + erff(g * 0.70710678118654752f));
}

// LDS byte offset of a __shared__ object (addrspace(3) ptrtoint)
__device__ __forceinline__ unsigned lds_off(const void* p) {
  return (unsigned)(unsigned long long)(const __attribute__((address_space(3))) void*)p;
}

// CDNA5 async global->LDS copy (ASYNCcnt-tracked); per-lane addresses
__device__ __forceinline__ void async_ld_b128(unsigned lds_byte_off, const void* g) {
  asm volatile("global_load_async_to_lds_b128 %0, %1, off"
               :: "v"(lds_byte_off), "v"((unsigned long long)(size_t)g)
               : "memory");
}

__device__ __forceinline__ void wait_async0() {
#if __has_builtin(__builtin_amdgcn_s_wait_asynccnt)
  __builtin_amdgcn_s_wait_asynccnt(0);
#else
  asm volatile("s_wait_asynccnt 0x0" ::: "memory");
#endif
}

// ---------------- fp32 -> bf16 convert (x4 vectorized) ----------------
__global__ void k_f32_to_bf16v4(const float* __restrict__ in,
                                unsigned short* __restrict__ out, int n4) {
  int i = blockIdx.x * blockDim.x + threadIdx.x;
  if (i >= n4) return;
  float4 v = ((const float4*)in)[i];
  __align__(8) unsigned short o[4] = { f2bf(v.x), f2bf(v.y), f2bf(v.z), f2bf(v.w) };
  ((uint2*)out)[i] = *(const uint2*)o;
}

// ---------------- gating: logits, softmax, top-3 (1 wave / token) -------------
__global__ void k_gate(const float* __restrict__ x, const float* __restrict__ gw,
                       int* __restrict__ tki, float* __restrict__ tkw) {
  int wave = (blockIdx.x * blockDim.x + threadIdx.x) >> 5;
  int lane = threadIdx.x & 31;
  if (wave >= T_TOK) return;
  const float* xr = x + (size_t)wave * DD;
  float logit[EE];
#pragma unroll
  for (int e = 0; e < EE; ++e) {
    const float* w = gw + e * DD;
    float s = 0.f;
    for (int d = lane; d < DD; d += 32) s += xr[d] * w[d];
#pragma unroll
    for (int off = 16; off > 0; off >>= 1) s += __shfl_xor(s, off);
    logit[e] = s;
  }
  float mx = logit[0];
#pragma unroll
  for (int e = 1; e < EE; ++e) mx = fmaxf(mx, logit[e]);
  float p[EE]; float den = 0.f;
#pragma unroll
  for (int e = 0; e < EE; ++e) { p[e] = __expf(logit[e] - mx); den += p[e]; }
  float invd = 1.f / den;
#pragma unroll
  for (int e = 0; e < EE; ++e) p[e] *= invd;
  if (lane == 0) {
    int idx[KTOP]; float wv[KTOP]; float wsum = 0.f;
#pragma unroll
    for (int j = 0; j < KTOP; ++j) {
      int bi = 0; float bv = -1.f;
#pragma unroll
      for (int e = 0; e < EE; ++e) if (p[e] > bv) { bv = p[e]; bi = e; }
      idx[j] = bi; wv[j] = bv; p[bi] = -2.f; wsum += bv;
    }
    float inv = 1.f / (wsum + 1e-20f);
#pragma unroll
    for (int j = 0; j < KTOP; ++j) {
      tki[wave * KTOP + j] = idx[j];
      tkw[wave * KTOP + j] = wv[j] * inv;
    }
  }
}

// ---------------- dispatch: counts -> 128-padded bases -> row lists -----------
__global__ void k_dispatch(const int* __restrict__ tki, const float* __restrict__ tkw,
                           int* __restrict__ bases, int* __restrict__ row_tok,
                           float* __restrict__ row_w, int* __restrict__ row_e) {
  __shared__ int cnt[EE];
  __shared__ int base_s[EE + 1];
  __shared__ int slot[EE];
  int tid = threadIdx.x;
  if (tid < EE) { cnt[tid] = 0; slot[tid] = 0; }
  __syncthreads();
  for (int t = tid; t < T_TOK; t += blockDim.x)
    for (int j = 0; j < KTOP; ++j)
      atomicAdd(&cnt[tki[t * KTOP + j]], 1);
  __syncthreads();
  if (tid == 0) {
    int off = 0;
    for (int e = 0; e < EE; ++e) {
      base_s[e] = off;
      off += (cnt[e] + 127) & ~127;   // pad to 128 = GEMM BM tile
    }
    base_s[EE] = off;
    for (int e = 0; e <= EE; ++e) bases[e] = base_s[e];
    for (int t = 0; t < T_TOK; ++t)   // serial: deterministic ordering
      for (int j = 0; j < KTOP; ++j) {
        int e = tki[t * KTOP + j];
        int pos = base_s[e] + slot[e]++;
        row_tok[pos] = t; row_w[pos] = tkw[t * KTOP + j]; row_e[pos] = e;
      }
    for (int e = 0; e < EE; ++e)
      for (int pos = base_s[e] + cnt[e]; pos < base_s[e + 1]; ++pos) {
        row_tok[pos] = 0; row_w[pos] = 0.f; row_e[pos] = e;
      }
  }
}

// ======================= WMMA GEMM core =======================
// C[m,n] = sum_k A[m,k]*B[n,k] (+bias[n]); A,B bf16 K-contiguous.
// Block 256 thr = 8 waves (4M x 2N); block tile 128M x 128N; wave tile 32M x 64N.
// Stage depth BK=64 (two 32-wide WMMA chunks per sync), double-buffered LDS
// fed by async global->LDS (ASYNCcnt). Persistent per-thread prefetch pointers.
#define GEMM_SETUP()                                                           \
  __shared__ __align__(16) unsigned short As[2][128][LDSP];                    \
  __shared__ __align__(16) unsigned short Bs[2][128][LDSP];                    \
  int tid = threadIdx.x;                                                       \
  int wid = tid >> 5, lane = tid & 31;                                         \
  int wm = wid & 3, wn = wid >> 2;                                             \
  int half = lane >> 4, mrow = lane & 15, koff = half * 8;                     \
  int rA = tid >> 3, c8w = (tid & 7) * 8;  /* staging: 4 rows x 1 b128 chunk */\
  v8f acc[2][4];                                                               \
  {                                                                            \
    v8f vz = {};                                                               \
    _Pragma("unroll") for (int mi = 0; mi < 2; ++mi)                           \
      _Pragma("unroll") for (int ni = 0; ni < 4; ++ni) acc[mi][ni] = vz;       \
  }

// issue 8 async b128s per thread (A 128x64 + B 128x64); bump pointers by BK
#define ISSUE_STAGE(bufv)                                                      \
  {                                                                            \
    _Pragma("unroll") for (int j = 0; j < 4; ++j) {                            \
      async_ld_b128(lds_off(&As[bufv][rA + j * 32][c8w]), pa[j]);              \
      async_ld_b128(lds_off(&Bs[bufv][rA + j * 32][c8w]), pb[j]);              \
      pa[j] += 64; pb[j] += 64;                                                \
    }                                                                          \
  }

// per-wave: 2 chunks x (2 A frags + 4 B frags -> 8 WMMAs) = 16 WMMAs per stage
#define COMPUTE_STAGE(bufv)                                                    \
  _Pragma("unroll") for (int kc = 0; kc < 64; kc += 32) {                      \
    __align__(32) unsigned short fa[2][16], fb[4][16];                         \
    _Pragma("unroll") for (int mi = 0; mi < 2; ++mi) {                         \
      const unsigned short* s = &As[bufv][wm * 32 + mi * 16 + mrow][kc + koff];\
      *(uint4*)&fa[mi][0] = *(const uint4*)s;                                  \
      *(uint4*)&fa[mi][8] = *(const uint4*)(s + 16);                           \
    }                                                                          \
    _Pragma("unroll") for (int ni = 0; ni < 4; ++ni) {                         \
      const unsigned short* s = &Bs[bufv][wn * 64 + ni * 16 + mrow][kc + koff];\
      *(uint4*)&fb[ni][0] = *(const uint4*)s;                                  \
      *(uint4*)&fb[ni][8] = *(const uint4*)(s + 16);                           \
    }                                                                          \
    _Pragma("unroll") for (int mi = 0; mi < 2; ++mi)                           \
      _Pragma("unroll") for (int ni = 0; ni < 4; ++ni)                         \
        acc[mi][ni] = __builtin_amdgcn_wmma_f32_16x16x32_bf16(                 \
            false, *(const v16bf*)fa[mi], false, *(const v16bf*)fb[ni],        \
            (short)0, acc[mi][ni], false, false);                              \
  }

#define GEMM_MAIN_LOOP(Kval)                                                   \
  ISSUE_STAGE(0);                                                              \
  wait_async0();                                                               \
  __syncthreads();                                                             \
  int buf = 0;                                                                 \
  for (int k0 = 0; k0 < (Kval); k0 += 64) {                                    \
    if (k0 + 64 < (Kval)) ISSUE_STAGE(buf ^ 1);                                \
    COMPUTE_STAGE(buf);                                                        \
    wait_async0();                                                             \
    __syncthreads();                                                           \
    buf ^= 1;                                                                  \
  }

// ---------------- generic dense GEMM (shared-expert path) ----------------
__global__ void __launch_bounds__(256)
k_gemm_nt(const unsigned short* __restrict__ A, const unsigned short* __restrict__ B,
          const float* __restrict__ bias, float* __restrict__ C,
          int M, int N, int K) {
  (void)M;
  int m0 = blockIdx.y * 128, n0 = blockIdx.x * 128;
  GEMM_SETUP();
  const unsigned short* pa[4]; const unsigned short* pb[4];
#pragma unroll
  for (int j = 0; j < 4; ++j) {
    pa[j] = A + (size_t)(m0 + rA + j * 32) * K + c8w;
    pb[j] = B + (size_t)(n0 + rA + j * 32) * K + c8w;
  }
  GEMM_MAIN_LOOP(K);
#pragma unroll
  for (int mi = 0; mi < 2; ++mi)
#pragma unroll
    for (int ni = 0; ni < 4; ++ni) {
      int col = n0 + wn * 64 + ni * 16 + mrow;
      float bc = bias[col];
#pragma unroll
      for (int i = 0; i < 8; ++i) {
        int row = m0 + wm * 32 + mi * 16 + half * 8 + i;
        C[(size_t)row * N + col] = acc[mi][ni][i] + bc;
      }
    }
}

// ---------------- expert fc1: gathered A rows, per-expert W/bias ----------
__global__ void __launch_bounds__(256)
k_expert_fc1(const unsigned short* __restrict__ Xb, const unsigned short* __restrict__ W1,
             const float* __restrict__ B1, const int* __restrict__ bases,
             const int* __restrict__ row_tok, float* __restrict__ Hp) {
  int m0 = blockIdx.y * 128;
  if (m0 >= bases[EE]) return;
  int e = 0;
  while (m0 >= bases[e + 1]) ++e;                 // uniform per block
  const unsigned short* W = W1 + (size_t)e * (2 * FF) * DD;
  const float* bias = B1 + (size_t)e * (2 * FF);
  int n0 = blockIdx.x * 128;
  GEMM_SETUP();
  const unsigned short* pa[4]; const unsigned short* pb[4];
#pragma unroll
  for (int j = 0; j < 4; ++j) {
    pa[j] = Xb + (size_t)row_tok[m0 + rA + j * 32] * DD + c8w;
    pb[j] = W + (size_t)(n0 + rA + j * 32) * DD + c8w;
  }
  GEMM_MAIN_LOOP(DD);
#pragma unroll
  for (int mi = 0; mi < 2; ++mi)
#pragma unroll
    for (int ni = 0; ni < 4; ++ni) {
      int col = n0 + wn * 64 + ni * 16 + mrow;
      float bc = bias[col];
#pragma unroll
      for (int i = 0; i < 8; ++i) {
        int row = m0 + wm * 32 + mi * 16 + half * 8 + i;
        Hp[(size_t)row * (2 * FF) + col] = acc[mi][ni][i] + bc;
      }
    }
}

// ---------------- expert fc2: scatter w*(acc+bias) via float atomics ------
__global__ void __launch_bounds__(256)
k_expert_fc2(const unsigned short* __restrict__ Ap, const unsigned short* __restrict__ W2,
             const float* __restrict__ B2, const int* __restrict__ bases,
             const int* __restrict__ row_tok, const float* __restrict__ row_w,
             float* __restrict__ out) {
  int m0 = blockIdx.y * 128;
  if (m0 >= bases[EE]) return;
  int e = 0;
  while (m0 >= bases[e + 1]) ++e;
  const unsigned short* W = W2 + (size_t)e * DD * FF;
  const float* bias = B2 + (size_t)e * DD;
  int n0 = blockIdx.x * 128;
  GEMM_SETUP();
  const unsigned short* pa[4]; const unsigned short* pb[4];
#pragma unroll
  for (int j = 0; j < 4; ++j) {
    pa[j] = Ap + (size_t)(m0 + rA + j * 32) * FF + c8w;
    pb[j] = W + (size_t)(n0 + rA + j * 32) * FF + c8w;
  }
  GEMM_MAIN_LOOP(FF);
#pragma unroll
  for (int mi = 0; mi < 2; ++mi) {
    int rbase = m0 + wm * 32 + mi * 16 + half * 8;
    int   tok[8]; float wv[8];
#pragma unroll
    for (int i = 0; i < 8; ++i) { tok[i] = row_tok[rbase + i]; wv[i] = row_w[rbase + i]; }
#pragma unroll
    for (int ni = 0; ni < 4; ++ni) {
      int col = n0 + wn * 64 + ni * 16 + mrow;
      float bc = bias[col];
#pragma unroll
      for (int i = 0; i < 8; ++i)
        atomicAdd(&out[(size_t)tok[i] * DD + col], wv[i] * (acc[mi][ni][i] + bc));
    }
  }
}

// ---------------- GEGLU (exact gelu) ----------------
__global__ void k_geglu_shared(const float* __restrict__ hs, const float* __restrict__ mult,
                               unsigned short* __restrict__ as_) {
  int i = blockIdx.x * blockDim.x + threadIdx.x;
  if (i >= T_TOK * FSH) return;
  int t = i / FSH, f = i - t * FSH;
  float xh = hs[(size_t)t * (2 * FSH) + f];
  float g  = hs[(size_t)t * (2 * FSH) + FSH + f];
  as_[i] = f2bf(gelu_exact(g) * xh * mult[f]);
}

__global__ void k_geglu_expert(const float* __restrict__ Hp, const float* __restrict__ mult,
                               const int* __restrict__ row_e, const int* __restrict__ bases,
                               unsigned short* __restrict__ Ap) {
  int i = blockIdx.x * blockDim.x + threadIdx.x;
  int r = i / FF, f = i - r * FF;
  if (r >= bases[EE]) return;
  float xh = Hp[(size_t)r * (2 * FF) + f];
  float g  = Hp[(size_t)r * (2 * FF) + FF + f];
  Ap[(size_t)r * FF + f] = f2bf(gelu_exact(g) * xh * mult[(size_t)row_e[r] * FF + f]);
}

// ---------------- host launcher ----------------
extern "C" void kernel_launch(void* const* d_in, const int* in_sizes, int n_in,
                              void* d_out, int out_size, void* d_ws, size_t ws_size,
                              hipStream_t stream) {
  (void)in_sizes; (void)n_in; (void)out_size; (void)ws_size;
  const float* x        = (const float*)d_in[0];
  const float* gate_w   = (const float*)d_in[1];
  const float* fc1_w    = (const float*)d_in[2];
  const float* fc1_b    = (const float*)d_in[3];
  const float* gmult    = (const float*)d_in[4];
  const float* fc2_w    = (const float*)d_in[5];
  const float* fc2_b    = (const float*)d_in[6];
  const float* s_fc1_w  = (const float*)d_in[7];
  const float* s_fc1_b  = (const float*)d_in[8];
  const float* s_gmult  = (const float*)d_in[9];
  const float* s_fc2_w  = (const float*)d_in[10];
  const float* s_fc2_b  = (const float*)d_in[11];
  float* out = (float*)d_out;

  char* ws = (char*)d_ws;
  size_t off = 0;
  auto alloc = [&](size_t bytes) -> char* {
    off = (off + 255) & ~(size_t)255;
    char* p = ws + off; off += bytes; return p;
  };
  unsigned short* x_bf     = (unsigned short*)alloc((size_t)T_TOK * DD * 2);
  unsigned short* sfc1w_bf = (unsigned short*)alloc((size_t)2 * FSH * DD * 2);
  unsigned short* sfc2w_bf = (unsigned short*)alloc((size_t)DD * FSH * 2);
  unsigned short* fc1w_bf  = (unsigned short*)alloc((size_t)EE * 2 * FF * DD * 2);
  unsigned short* fc2w_bf  = (unsigned short*)alloc((size_t)EE * DD * FF * 2);
  int*   tki     = (int*)alloc((size_t)T_TOK * KTOP * 4);
  float* tkw     = (float*)alloc((size_t)T_TOK * KTOP * 4);
  int*   bases   = (int*)alloc((EE + 1) * 4);
  int*   row_tok = (int*)alloc((size_t)POOL_CAP * 4);
  float* row_w   = (float*)alloc((size_t)POOL_CAP * 4);
  int*   row_e   = (int*)alloc((size_t)POOL_CAP * 4);
  unsigned short* a_s = (unsigned short*)alloc((size_t)T_TOK * FSH * 2);
  unsigned short* a_p = (unsigned short*)alloc((size_t)POOL_CAP * FF * 2);
  // union region: hs (T x 2FSH f32) then h_pool (POOL_CAP x 2FF f32); disjoint lifetimes
  size_t hs_bytes = (size_t)T_TOK * 2 * FSH * 4;
  size_t hp_bytes = (size_t)POOL_CAP * 2 * FF * 4;
  char* un = alloc(hs_bytes > hp_bytes ? hs_bytes : hp_bytes);
  float* hs = (float*)un;
  float* hp = (float*)un;

  // 1) fp32 -> bf16 conversions
  auto cvt = [&](const float* src, unsigned short* dst, size_t n) {
    size_t n4 = n / 4;
    k_f32_to_bf16v4<<<(unsigned)((n4 + 255) / 256), 256, 0, stream>>>(src, dst, (int)n4);
  };
  cvt(x,       x_bf,     (size_t)T_TOK * DD);
  cvt(s_fc1_w, sfc1w_bf, (size_t)2 * FSH * DD);
  cvt(s_fc2_w, sfc2w_bf, (size_t)DD * FSH);
  cvt(fc1_w,   fc1w_bf,  (size_t)EE * 2 * FF * DD);
  cvt(fc2_w,   fc2w_bf,  (size_t)EE * DD * FF);

  // 2) gating
  k_gate<<<T_TOK / 8, 256, 0, stream>>>(x, gate_w, tki, tkw);

  // 3) dispatch
  k_dispatch<<<1, 256, 0, stream>>>(tki, tkw, bases, row_tok, row_w, row_e);

  // 4) shared fc1: (T x 2FSH) = x_bf @ s_fc1_w^T + b
  k_gemm_nt<<<dim3(2 * FSH / 128, T_TOK / 128), 256, 0, stream>>>(
      x_bf, sfc1w_bf, s_fc1_b, hs, T_TOK, 2 * FSH, DD);

  // 5) shared GEGLU -> a_s (bf16)
  k_geglu_shared<<<(T_TOK * FSH) / 256, 256, 0, stream>>>(hs, s_gmult, a_s);

  // 6) shared fc2: writes full d_out (plain stores)
  k_gemm_nt<<<dim3(DD / 128, T_TOK / 128), 256, 0, stream>>>(
      a_s, sfc2w_bf, s_fc2_b, out, T_TOK, DD, FSH);

  // 7) expert fc1 (gathered rows) -> h_pool (reuses hs region)
  k_expert_fc1<<<dim3(2 * FF / 128, POOL_CAP / 128), 256, 0, stream>>>(
      x_bf, fc1w_bf, fc1_b, bases, row_tok, hp);

  // 8) expert GEGLU -> a_pool (bf16)
  k_geglu_expert<<<(unsigned)(((size_t)POOL_CAP * FF) / 256), 256, 0, stream>>>(
      hp, gmult, row_e, bases, a_p);

  // 9) expert fc2: scatter-add w*(acc+bias) into d_out
  k_expert_fc2<<<dim3(DD / 128, POOL_CAP / 128), 256, 0, stream>>>(
      a_p, fc2w_bf, fc2_b, bases, row_tok, row_w, out);
}
